// TriangleSelfAttentionRowWise_22574348108023
// MI455X (gfx1250) — matrix operand order repaired and checked by reading, not compiled, not run on hardware
//
#include <hip/hip_runtime.h>
#include <hip/hip_bf16.h>

typedef _Float16 half_t;
typedef __attribute__((ext_vector_type(16))) _Float16 v16h;
typedef __attribute__((ext_vector_type(8)))  _Float16 v8h;
typedef __attribute__((ext_vector_type(8)))  float    v8f;

#define EDIM 128
#define JDIM 256
#define HNUM 4
#define CDIM 32
#define LN_EPS 1e-5f

// ---------------- CDNA5 async global->LDS helpers ----------------------------
// GLOBAL_LOAD_ASYNC_TO_LDS_B128 (GV mode): per-lane copy of 16B from global to
// LDS, tracked by ASYNCcnt. LDS address = low 32 bits of the generic pointer
// (ISA 10.2).
__device__ __forceinline__ void async_ld_b128(void* lds_ptr, const void* gptr) {
    unsigned lds = (unsigned)(unsigned long long)lds_ptr;
    asm volatile("global_load_async_to_lds_b128 %0, %1, off"
                 :: "v"(lds), "v"(gptr)
                 : "memory");
}
__device__ __forceinline__ void wait_async0() {
    asm volatile("s_wait_asynccnt 0x0" ::: "memory");
}

// ---------------- WMMA fragment helpers (CDNA5 16x16x32 f16, wave32) --------

// A fragment: 16(M) x 32(K) f16. Lane l: m = l&15, k-base = (l>=16)*8.
// halfs 0..7 -> k = kb+0..7 ; halfs 8..15 -> k = kb+16..23. Two b128 loads.
__device__ __forceinline__ v16h load_frag_a(const half_t* __restrict__ src,
                                            int row0, int k0, int stride, int lane) {
    int m  = lane & 15;
    int kb = (lane >> 4) << 3;
    const half_t* p = src + (size_t)(row0 + m) * stride + k0 + kb;
    v8h lo = *(const v8h*)p;
    v8h hi = *(const v8h*)(p + 16);
    return __builtin_shufflevector(lo, hi, 0,1,2,3,4,5,6,7,8,9,10,11,12,13,14,15);
}

// B fragment: 32(K) x 16(N) f16 from K-contiguous storage src[n*stride + k].
// Lane l: n = l&15, k-base = (l>=16)*8; halfs 0..7 -> kb+0..7, 8..15 -> kb+16..23. Two b128 loads.
__device__ __forceinline__ v16h load_frag_b(const half_t* __restrict__ src,
                                            int n0, int k0, int stride, int lane) {
    int n  = lane & 15;
    int kb = (lane >> 4) << 3;
    const half_t* p = src + (size_t)(n0 + n) * stride + k0 + kb;
    v8h lo = *(const v8h*)p;
    v8h hi = *(const v8h*)(p + 16);
    return __builtin_shufflevector(lo, hi, 0,1,2,3,4,5,6,7,8,9,10,11,12,13,14,15);
}

__device__ __forceinline__ v8f wmma_f16(v16h a, v16h b, v8f c) {
    return __builtin_amdgcn_wmma_f32_16x16x32_f16(false, a, false, b, (short)0, c, false, false);
}

// C/D layout: lane l holds col n = l&15, rows m = e + (l>=16)*8, e = 0..7.

// ---------------- Kernel 1: LayerNorm + Q/K/V/Gate projections ---------------
// One workgroup per i (256 WGs x 256 threads = 8 waves).
// Outputs q,k,v,g as f16 in [i][h][j][c] layout.
__global__ __launch_bounds__(256) void k_ln_qkvg(
    const float* __restrict__ z, const float* __restrict__ lscale, const float* __restrict__ lbias,
    const float* __restrict__ Wq, const float* __restrict__ Wk,
    const float* __restrict__ Wv, const float* __restrict__ Wg,
    const float* __restrict__ bg,
    half_t* __restrict__ qout, half_t* __restrict__ kout,
    half_t* __restrict__ vout, half_t* __restrict__ gout)
{
    __shared__ __align__(16) half_t zn[JDIM * EDIM];   // 64 KB, row-major [j][e]
    __shared__ __align__(16) half_t Wt[EDIM * EDIM];   // 32 KB, transposed [out][in]

    const int i    = blockIdx.x;
    const int tid  = threadIdx.x;
    const int lane = tid & 31;
    const int wv   = tid >> 5;

    // --- LayerNorm: one token per thread (E = 128) ---
    {
        const float4* z4 = (const float4*)(z + ((size_t)i * JDIM + tid) * EDIM);
        float s = 0.f, s2 = 0.f;
        for (int e = 0; e < 32; e++) {
            float4 x = z4[e];
            s  += x.x + x.y + x.z + x.w;
            s2 += x.x*x.x + x.y*x.y + x.z*x.z + x.w*x.w;
        }
        float mu  = s * (1.f / EDIM);
        float var = s2 * (1.f / EDIM) - mu * mu;
        float inv = rsqrtf(var + LN_EPS);
        for (int e = 0; e < 32; e++) {
            float4 x = z4[e];
            float4 sc = ((const float4*)lscale)[e];
            float4 bi = ((const float4*)lbias)[e];
            half_t* d = zn + tid * EDIM + e * 4;
            d[0] = (half_t)((x.x - mu) * inv * sc.x + bi.x);
            d[1] = (half_t)((x.y - mu) * inv * sc.y + bi.y);
            d[2] = (half_t)((x.z - mu) * inv * sc.z + bi.z);
            d[3] = (half_t)((x.w - mu) * inv * sc.w + bi.w);
        }
    }

    // --- Four projections, one weight matrix at a time ---
    for (int wsel = 0; wsel < 4; wsel++) {
        const float* W   = (wsel == 0) ? Wq : (wsel == 1) ? Wk : (wsel == 2) ? Wv : Wg;
        half_t*      dst = (wsel == 0) ? qout : (wsel == 1) ? kout : (wsel == 2) ? vout : gout;

        __syncthreads();   // zn ready (iter 0) / previous GEMM reads of Wt done
        // transpose-convert W[k][n] (f32) -> Wt[n][k] (f16)
        for (int t = tid; t < EDIM * EDIM; t += 256) {
            int n = t >> 7, k = t & 127;
            Wt[t] = (half_t)W[k * EDIM + n];
        }
        __syncthreads();

        // 256x128 = 16 Mtiles x 8 Ntiles; wave w owns column nt = w -> B frags
        // are loop-invariant per wave (hoisted into VGPRs by the compiler).
        for (int tile = wv; tile < 16 * 8; tile += 8) {
            int mt = tile >> 3, nt = tile & 7;
            v8f acc = {};
            #pragma unroll
            for (int ks = 0; ks < 4; ks++) {
                v16h a = load_frag_a(zn, mt * 16, ks * 32, EDIM, lane);
                v16h b = load_frag_b(Wt, nt * 16, ks * 32, EDIM, lane);
                acc = wmma_f16(a, b, acc);
            }
            int n = nt * 16 + (lane & 15);
            int h = n >> 5, c = n & 31;
            if (wsel == 3) {
                float bias = bg[n];
                #pragma unroll
                for (int e = 0; e < 8; e++) {
                    int j = mt * 16 + e + ((lane >> 4) << 3);
                    float x = acc[e] + bias;
                    float gsig = 1.f / (1.f + __expf(-x));
                    dst[(((size_t)i * HNUM + h) * JDIM + j) * CDIM + c] = (half_t)gsig;
                }
            } else {
                #pragma unroll
                for (int e = 0; e < 8; e++) {
                    int j = mt * 16 + e + ((lane >> 4) << 3);
                    dst[(((size_t)i * HNUM + h) * JDIM + j) * CDIM + c] = (half_t)acc[e];
                }
            }
        }
    }
}

// ---------------- Kernel 2: per-(i,h) row-wise attention + gate --------------
// grid (256, 4), 256 threads. logits/softmax in 64-row q blocks; q blocks are
// double-buffered with async global->LDS copies overlapping compute.
__global__ __launch_bounds__(256) void k_attn(
    const half_t* __restrict__ qg, const half_t* __restrict__ kg,
    const half_t* __restrict__ vg, const half_t* __restrict__ gg,
    half_t* __restrict__ att)
{
    __shared__ __align__(16) half_t kb[JDIM * CDIM];    // [jk][c]  16 KB
    __shared__ __align__(16) half_t vT[CDIM * JDIM];    // [c][jk]  16 KB
    __shared__ __align__(16) half_t qb[2][64 * CDIM];   //           8 KB
    __shared__ __align__(16) float  lg[64 * JDIM];      // logits   64 KB
    __shared__ __align__(16) half_t wb[64 * JDIM];      // exp(l-m) 32 KB
    __shared__ float red_m[64 * 4];
    __shared__ float red_s[64 * 4];

    const int i    = blockIdx.x;
    const int h    = blockIdx.y;
    const int tid  = threadIdx.x;
    const int lane = tid & 31;
    const int wv   = tid >> 5;
    const size_t base = ((size_t)i * HNUM + h) * JDIM * CDIM;

    // async stage K (1024 b128, 4 per thread)
    for (int t = tid; t < JDIM * CDIM / 8; t += 256)
        async_ld_b128(kb + t * 8, kg + base + (size_t)t * 8);
    // async stage q block 0 (256 b128, 1 per thread)
    for (int t = tid; t < 64 * CDIM / 8; t += 256)
        async_ld_b128(qb[0] + t * 8, qg + base + (size_t)t * 8);
    // stage V transposed (manual scatter): thread t owns row j = t
    {
        const half_t* vr = vg + base + (size_t)tid * CDIM;
        for (int c = 0; c < CDIM; c++) vT[c * JDIM + tid] = vr[c];
    }
    wait_async0();
    __syncthreads();

    for (int qbk = 0; qbk < 4; qbk++) {
        const half_t* qcur = qb[qbk & 1];
        // prefetch next q block asynchronously; overlaps with this block's work
        if (qbk < 3) {
            half_t* qnext = qb[(qbk + 1) & 1];
            const half_t* src = qg + base + (size_t)(qbk + 1) * 64 * CDIM;
            for (int t = tid; t < 64 * CDIM / 8; t += 256)
                async_ld_b128(qnext + t * 8, src + (size_t)t * 8);
        }

        // logits: 4 Mtiles x 16 Ntiles, K = 32 -> one WMMA per tile
        for (int tile = wv; tile < 64; tile += 8) {
            int mt = tile >> 4, nt = tile & 15;
            v16h a = load_frag_a(qcur, mt * 16, 0, CDIM, lane);
            v16h b = load_frag_b(kb, nt * 16, 0, CDIM, lane);
            v8f acc = {};
            acc = wmma_f16(a, b, acc);
            int col = nt * 16 + (lane & 15);
            #pragma unroll
            for (int e = 0; e < 8; e++) {
                int row = mt * 16 + e + ((lane >> 4) << 3);
                lg[row * JDIM + col] = acc[e];
            }
        }
        __syncthreads();

        // softmax: 4 threads per row, 64-elem segments; unnormalized exp into wb
        {
            int r = tid >> 2, seg = tid & 3;
            const float* lr = lg + r * JDIM + seg * 64;
            float mx = -1e30f;
            for (int e = 0; e < 64; e++) mx = fmaxf(mx, lr[e]);
            red_m[r * 4 + seg] = mx;
        }
        __syncthreads();
        {
            int r = tid >> 2, seg = tid & 3;
            float mx = fmaxf(fmaxf(red_m[r * 4 + 0], red_m[r * 4 + 1]),
                             fmaxf(red_m[r * 4 + 2], red_m[r * 4 + 3]));
            const float* lr = lg + r * JDIM + seg * 64;
            half_t* wr = wb + r * JDIM + seg * 64;
            float s = 0.f;
            for (int e = 0; e < 64; e++) {
                float p = __expf(lr[e] - mx);
                s += p;
                wr[e] = (half_t)p;
            }
            red_s[r * 4 + seg] = s;
        }
        __syncthreads();

        // wavg: 4 Mtiles x 2 Ntiles = 8 tiles, one per wave; K = 256 -> 8 WMMAs
        {
            int mt = wv >> 1, nt = wv & 1;
            v8f acc = {};
            #pragma unroll
            for (int ks = 0; ks < 8; ks++) {
                v16h a = load_frag_a(wb, mt * 16, ks * 32, JDIM, lane);
                v16h b = load_frag_b(vT, nt * 16, ks * 32, JDIM, lane);
                acc = wmma_f16(a, b, acc);
            }
            int c = nt * 16 + (lane & 15);
            #pragma unroll
            for (int e = 0; e < 8; e++) {
                int rl = mt * 16 + e + ((lane >> 4) << 3);
                int jq = qbk * 64 + rl;
                float s = red_s[rl * 4 + 0] + red_s[rl * 4 + 1] +
                          red_s[rl * 4 + 2] + red_s[rl * 4 + 3];
                float val = acc[e] / s;
                float gv = (float)gg[base + (size_t)jq * CDIM + c];
                att[((size_t)i * JDIM + jq) * (HNUM * CDIM) + h * CDIM + c] = (half_t)(val * gv);
            }
        }
        wait_async0();      // next q block resident before reuse
        __syncthreads();
    }
}

// ---------------- Kernel 3: output projection + bias -------------------------
__global__ __launch_bounds__(256) void k_out(
    const half_t* __restrict__ att, const float* __restrict__ Wo,
    const float* __restrict__ bo, float* __restrict__ out)
{
    __shared__ __align__(16) half_t ab[256 * EDIM];   // 64 KB
    __shared__ __align__(16) half_t Wt[EDIM * EDIM];  // 32 KB

    const int tid  = threadIdx.x;
    const int lane = tid & 31;
    const int wv   = tid >> 5;
    const size_t row0 = (size_t)blockIdx.x * 256;

    // async stage 256 activation rows (4096 b128, 16 per thread)
    for (int t = tid; t < 256 * EDIM / 8; t += 256)
        async_ld_b128(ab + t * 8, att + row0 * EDIM + (size_t)t * 8);
    // transpose-convert Wo[k][n] (f32) -> Wt[n][k] (f16)
    for (int t = tid; t < EDIM * EDIM; t += 256) {
        int n = t >> 7, k = t & 127;
        Wt[t] = (half_t)Wo[k * EDIM + n];
    }
    wait_async0();
    __syncthreads();

    for (int tile = wv; tile < 16 * 8; tile += 8) {
        int mt = tile >> 3, nt = tile & 7;
        v8f acc = {};
        #pragma unroll
        for (int ks = 0; ks < 4; ks++) {
            v16h a = load_frag_a(ab, mt * 16, ks * 32, EDIM, lane);
            v16h b = load_frag_b(Wt, nt * 16, ks * 32, EDIM, lane);
            acc = wmma_f16(a, b, acc);
        }
        int n = nt * 16 + (lane & 15);
        float bias = bo[n];
        #pragma unroll
        for (int e = 0; e < 8; e++) {
            int r = mt * 16 + e + ((lane >> 4) << 3);
            out[(row0 + r) * EDIM + n] = acc[e] + bias;
        }
    }
}

// ---------------- Launch ------------------------------------------------------
extern "C" void kernel_launch(void* const* d_in, const int* in_sizes, int n_in,
                              void* d_out, int out_size, void* d_ws, size_t ws_size,
                              hipStream_t stream) {
    const float* z  = (const float*)d_in[0];
    const float* ls = (const float*)d_in[1];
    const float* lb = (const float*)d_in[2];
    const float* Wq = (const float*)d_in[3];
    const float* Wk = (const float*)d_in[4];
    const float* Wv = (const float*)d_in[5];
    const float* Wg = (const float*)d_in[6];
    const float* bg = (const float*)d_in[7];
    const float* Wo = (const float*)d_in[8];
    const float* bo = (const float*)d_in[9];
    float* out = (float*)d_out;

    const size_t MAT = (size_t)256 * 256 * 128;   // 8,388,608 elements
    half_t* q   = (half_t*)d_ws;
    half_t* k   = q + MAT;
    half_t* v   = k + MAT;
    half_t* g   = v + MAT;
    half_t* att = g + MAT;                         // total 5*MAT*2 = 80 MiB

    k_ln_qkvg<<<dim3(256), dim3(256), 0, stream>>>(z, ls, lb, Wq, Wk, Wv, Wg, bg, q, k, v, g);
    k_attn   <<<dim3(256, 4), dim3(256), 0, stream>>>(q, k, v, g, att);
    k_out    <<<dim3(256), dim3(256), 0, stream>>>(att, Wo, bo, out);
}
